// MultiHeadAttention_82824149336918
// MI455X (gfx1250) — compile-verified, matrix operand order checked by reading, not run on hardware
//
#include <hip/hip_runtime.h>
#include <stdint.h>

typedef __bf16 bf16;
typedef __attribute__((ext_vector_type(16))) __bf16 v16bf;
typedef __attribute__((ext_vector_type(8)))  float  v8f;

#define B_  2
#define S_  2048
#define D_  1024
#define H_  16
#define DH_ 64

// ---------------------------------------------------------------------------
// Fragment loaders (CDNA5 WMMA 16x16x32 bf16 layouts, ISA 7.12.2, wave32)
// ---------------------------------------------------------------------------
__device__ inline v16bf frag_rowmajor(const bf16* t, int ld, int koff) {
    int lane = threadIdx.x & 31;
    int m    = lane & 15;
    int kb   = (lane < 16) ? 0 : 8;
    const bf16* rp = t + m * ld + koff;
    v16bf f;
#pragma unroll
    for (int i = 0; i < 8; ++i) f[i]     = rp[kb + i];
#pragma unroll
    for (int i = 0; i < 8; ++i) f[8 + i] = rp[16 + kb + i];
    return f;
}

// B fragment from a K-major tile (tile[k][n]): B^T[n][k] = tile[k*ld + nbase + n]
__device__ inline v16bf frag_kmajor(const bf16* t, int ld, int nbase) {
    int lane = threadIdx.x & 31;
    int n    = lane & 15;
    int kb   = (lane < 16) ? 0 : 8;
    const bf16* cp = t + nbase + n;
    v16bf f;
#pragma unroll
    for (int i = 0; i < 8; ++i) f[i]     = cp[(kb + i) * ld];
#pragma unroll
    for (int i = 0; i < 8; ++i) f[8 + i] = cp[(16 + kb + i) * ld];
    return f;
}

// ---------------------------------------------------------------------------
// gfx1250 async global->LDS copy (ASYNCcnt path), 16B per lane per issue.
// ---------------------------------------------------------------------------
__device__ inline void async_b128(uint32_t lds_byte, const bf16* g) {
    asm volatile("global_load_async_to_lds_b128 %0, %1, off"
                 :: "v"(lds_byte), "v"(g)
                 : "memory");
}
__device__ inline void wait_async_le2() {
    asm volatile("s_wait_asynccnt 0x2" ::: "memory");
}
__device__ inline void wait_async_0() {
    asm volatile("s_wait_asynccnt 0x0" ::: "memory");
}

// Issue one 32-row x 64-col bf16 tile (4 KB): 2 async b128 per thread (128 thr).
__device__ inline void issue_tile32(uint32_t ldsBase, const bf16* gtile) {
    const int tid = threadIdx.x;
#pragma unroll
    for (int i = 0; i < 2; ++i) {
        int idx = tid + i * 128;             // 256 chunks of 16B
        int r   = idx >> 3;                  // row 0..31
        int c8  = (idx & 7) * 8;             // col start (bf16 units)
        async_b128(ldsBase + (uint32_t)idx * 16u, gtile + (size_t)r * D_ + c8);
    }
}

// ---------------------------------------------------------------------------
// Projection GEMM: Out(bf16)[M,N] = X(f32)[M,K] * W(f32)[K,N] + bias
// (f32->bf16 conversion forces the VGPR-staged path; WGP-scope prefetch of
//  the next K-slab: locality 3 -> scope WGP, pulls into all cache levels)
// ---------------------------------------------------------------------------
__global__ __launch_bounds__(128)
void ProjGemmF32ToBf16(const float* __restrict__ X, const float* __restrict__ W,
                       const float* __restrict__ bias, bf16* __restrict__ Out) {
    __shared__ bf16 ldsA[64 * 32];      // [m][k]
    __shared__ bf16 ldsB[64 * 32];      // [n][k]  (B^T)
    const int tid   = threadIdx.x;
    const int wave  = tid >> 5;
    const int nBase = blockIdx.x * 64;
    const int mBase = blockIdx.y * 64;

    v8f acc[4] = {};
    for (int kk = 0; kk < D_; kk += 32) {
        if (kk + 32 < D_) {   // global_prefetch_b8 hints for next slab (WGP scope)
            __builtin_prefetch(X + (size_t)(mBase + (tid >> 1)) * D_ + kk + 32 + (tid & 1) * 16, 0, 3);
            __builtin_prefetch(W + (size_t)(kk + 32 + (tid >> 2)) * D_ + nBase + (tid & 3) * 16, 0, 3);
        }
#pragma unroll
        for (int i = 0; i < 16; ++i) {              // A tile 64x32
            int idx = tid + i * 128;
            int r = idx >> 5, c = idx & 31;
            ldsA[idx] = (bf16)X[(size_t)(mBase + r) * D_ + kk + c];
        }
#pragma unroll
        for (int i = 0; i < 16; ++i) {              // W tile -> B^T
            int idx = tid + i * 128;
            int n = idx & 63, k = idx >> 6;
            ldsB[n * 32 + k] = (bf16)W[(size_t)(kk + k) * D_ + nBase + n];
        }
        __syncthreads();
        v16bf a = frag_rowmajor(ldsA + wave * 16 * 32, 32, 0);
#pragma unroll
        for (int nt = 0; nt < 4; ++nt) {
            v16bf b = frag_rowmajor(ldsB + nt * 16 * 32, 32, 0);
            acc[nt] = __builtin_amdgcn_wmma_f32_16x16x32_bf16(
                false, a, false, b, (short)0, acc[nt], false, false);
        }
        __syncthreads();
    }
    const int lane  = tid & 31;
    const int n     = lane & 15;
    const int rbase = (lane < 16) ? 0 : 8;
#pragma unroll
    for (int nt = 0; nt < 4; ++nt) {
        float bv = bias[nBase + nt * 16 + n];
#pragma unroll
        for (int r = 0; r < 8; ++r) {
            int row = mBase + wave * 16 + rbase + r;
            Out[(size_t)row * D_ + nBase + nt * 16 + n] = (bf16)(acc[nt][r] + bv);
        }
    }
}

// ---------------------------------------------------------------------------
// Output projection: out(f32) = ctx(bf16) * Wo + bo   (NT stores, write-once)
// ---------------------------------------------------------------------------
__global__ __launch_bounds__(128)
void OutProjGemm(const bf16* __restrict__ X, const float* __restrict__ W,
                 const float* __restrict__ bias, float* __restrict__ Out) {
    __shared__ bf16 ldsA[64 * 32];
    __shared__ bf16 ldsB[64 * 32];
    const int tid   = threadIdx.x;
    const int wave  = tid >> 5;
    const int nBase = blockIdx.x * 64;
    const int mBase = blockIdx.y * 64;

    v8f acc[4] = {};
    for (int kk = 0; kk < D_; kk += 32) {
        if (kk + 32 < D_) {
            __builtin_prefetch(X + (size_t)(mBase + (tid >> 1)) * D_ + kk + 32 + (tid & 1) * 16, 0, 3);
            __builtin_prefetch(W + (size_t)(kk + 32 + (tid >> 2)) * D_ + nBase + (tid & 3) * 16, 0, 3);
        }
#pragma unroll
        for (int i = 0; i < 16; ++i) {
            int idx = tid + i * 128;
            int r = idx >> 5, c = idx & 31;
            ldsA[idx] = X[(size_t)(mBase + r) * D_ + kk + c];
        }
#pragma unroll
        for (int i = 0; i < 16; ++i) {
            int idx = tid + i * 128;
            int n = idx & 63, k = idx >> 6;
            ldsB[n * 32 + k] = (bf16)W[(size_t)(kk + k) * D_ + nBase + n];
        }
        __syncthreads();
        v16bf a = frag_rowmajor(ldsA + wave * 16 * 32, 32, 0);
#pragma unroll
        for (int nt = 0; nt < 4; ++nt) {
            v16bf b = frag_rowmajor(ldsB + nt * 16 * 32, 32, 0);
            acc[nt] = __builtin_amdgcn_wmma_f32_16x16x32_bf16(
                false, a, false, b, (short)0, acc[nt], false, false);
        }
        __syncthreads();
    }
    const int lane  = tid & 31;
    const int n     = lane & 15;
    const int rbase = (lane < 16) ? 0 : 8;
#pragma unroll
    for (int nt = 0; nt < 4; ++nt) {
        float bv = bias[nBase + nt * 16 + n];
#pragma unroll
        for (int r = 0; r < 8; ++r) {
            int row = mBase + wave * 16 + rbase + r;
            __builtin_nontemporal_store(acc[nt][r] + bv,
                &Out[(size_t)row * D_ + nBase + nt * 16 + n]);
        }
    }
}

// ---------------------------------------------------------------------------
// Attention: per (b,h,64 q-rows). Two-pass, materialized attn matrix.
// All K/V/Q tile movement via async global->LDS copies, double-buffered.
// ---------------------------------------------------------------------------
__global__ __launch_bounds__(128)
void AttnKernel(const bf16* __restrict__ Q, const bf16* __restrict__ Kb,
                const bf16* __restrict__ Vb, const int* __restrict__ mask,
                float* __restrict__ attn, bf16* __restrict__ ctx) {
    __shared__ bf16 ldsQ[64 * 64];        // 8 KB
    __shared__ bf16 kvbuf[2][32 * 64];    // 2 x 4 KB, double-buffered K / V
    __shared__ bf16 ldsP[4][16 * 32];     // 4 KB, per-wave P tile (A layout)
    const int tid   = threadIdx.x;
    const int wave  = tid >> 5;
    const int lane  = tid & 31;
    const int qBase = blockIdx.x * 64;
    const int h     = blockIdx.y;
    const int b     = blockIdx.z;
    const float scale = 0.125f;           // 1/sqrt(DH=64)
    const int   NT    = S_ / 32;          // 64 key tiles

    const bf16* Qtile = Q  + (size_t)(b * S_ + qBase) * D_ + h * DH_;
    const bf16* Kbase = Kb + (size_t)(b * S_) * D_ + h * DH_;
    const bf16* Vbase = Vb + (size_t)(b * S_) * D_ + h * DH_;
    const uint32_t ldsQoff = (uint32_t)(uint64_t)&ldsQ[0];
    const uint32_t kv0     = (uint32_t)(uint64_t)&kvbuf[0][0];
    const uint32_t kv1     = (uint32_t)(uint64_t)&kvbuf[1][0];

    // ---- async-load Q tile (64x64, 8 KB: 4 chunks/thread) ----
#pragma unroll
    for (int i = 0; i < 4; ++i) {
        int idx = tid + i * 128;          // 512 chunks of 16B
        int r = idx >> 3, c8 = (idx & 7) * 8;
        async_b128(ldsQoff + (uint32_t)idx * 16u, Qtile + (size_t)r * D_ + c8);
    }
    issue_tile32(kv0, Kbase);             // K tile 0
    wait_async_le2();                     // own Q chunks done; K0 may be in flight
    __syncthreads();                      // all waves' Q chunks done
    v16bf aq0 = frag_rowmajor(ldsQ + wave * 16 * 64, 64, 0);
    v16bf aq1 = frag_rowmajor(ldsQ + wave * 16 * 64, 64, 32);

    float m[8], s[8];
#pragma unroll
    for (int r = 0; r < 8; ++r) { m[r] = -1e30f; s[r] = 0.0f; }

    const int n     = lane & 15;
    const int rbase = (lane < 16) ? 0 : 8;
    float* attnRow0 = attn +
        ((size_t)(b * H_ + h) * S_ + qBase + wave * 16 + rbase) * S_;

    // ---- Pass 1: scores = QK^T*scale (masked), online (max,sum) ----
    for (int j = 0; j < NT; ++j) {
        const bf16* curK = kvbuf[j & 1];
        if (j + 1 < NT) {
            issue_tile32((j & 1) ? kv0 : kv1, Kbase + (size_t)(j + 1) * 32 * D_);
            wait_async_le2();             // in-order: oldest tile landed
        } else {
            wait_async_0();
        }
        __syncthreads();
#pragma unroll
        for (int t = 0; t < 2; ++t) {
            v16bf b0 = frag_rowmajor(curK + t * 16 * 64, 64, 0);
            v16bf b1 = frag_rowmajor(curK + t * 16 * 64, 64, 32);
            v8f sc = {};
            sc = __builtin_amdgcn_wmma_f32_16x16x32_bf16(false, aq0, false, b0, (short)0, sc, false, false);
            sc = __builtin_amdgcn_wmma_f32_16x16x32_bf16(false, aq1, false, b1, (short)0, sc, false, false);
            int col    = j * 32 + t * 16 + n;
            bool valid = mask[b * S_ + col] != 0;
#pragma unroll
            for (int r = 0; r < 8; ++r) {
                float x  = valid ? sc[r] * scale : -1e30f;
                float nm = fmaxf(m[r], x);
                s[r] = s[r] * __expf(m[r] - nm) + __expf(x - nm);
                m[r] = nm;
                attnRow0[(size_t)r * S_ + col] = x;   // re-read in pass 2: keep RT
            }
        }
        __syncthreads();                  // protect curK before next overwrite
    }

    // ---- Merge (max,sum) across the 16 lanes sharing each row ----
#pragma unroll
    for (int r = 0; r < 8; ++r) {
#pragma unroll
        for (int off = 1; off < 16; off <<= 1) {
            float mo = __shfl_xor(m[r], off, 32);
            float so = __shfl_xor(s[r], off, 32);
            float nm = fmaxf(m[r], mo);
            s[r] = s[r] * __expf(m[r] - nm) + so * __expf(mo - nm);
            m[r] = nm;
        }
    }
    float inv[8];
#pragma unroll
    for (int r = 0; r < 8; ++r) inv[r] = 1.0f / s[r];

    // ---- Pass 2: p = exp(s-max)/sum, write probs (NT), ctx = P @ V ----
    v8f cacc[4] = {};
    issue_tile32(kv0, Vbase);             // V tile 0
    for (int jj = 0; jj < NT; ++jj) {
        const bf16* curV = kvbuf[jj & 1];
        if (jj + 1 < NT)
            issue_tile32((jj & 1) ? kv0 : kv1, Vbase + (size_t)(jj + 1) * 32 * D_);
        // normalize while the async copy flies; scores are dead after this
        // read -> non-temporal load (don't retain 537 MB of dead lines in L2)
#pragma unroll
        for (int t = 0; t < 2; ++t) {
            int col = jj * 32 + t * 16 + n;
#pragma unroll
            for (int r = 0; r < 8; ++r) {
                float x = __builtin_nontemporal_load(&attnRow0[(size_t)r * S_ + col]);
                float p = __expf(x - m[r]) * inv[r];
                __builtin_nontemporal_store(p, &attnRow0[(size_t)r * S_ + col]);
                ldsP[wave][(rbase + r) * 32 + t * 16 + n] = (bf16)p;
            }
        }
        if (jj + 1 < NT) wait_async_le2(); else wait_async_0();
        __syncthreads();
        v16bf ap = frag_rowmajor(ldsP[wave], 32, 0);   // per-wave: no barrier needed
#pragma unroll
        for (int nt = 0; nt < 4; ++nt) {
            v16bf bvv = frag_kmajor(curV, 64, nt * 16);
            cacc[nt] = __builtin_amdgcn_wmma_f32_16x16x32_bf16(
                false, ap, false, bvv, (short)0, cacc[nt], false, false);
        }
        __syncthreads();
    }
#pragma unroll
    for (int nt = 0; nt < 4; ++nt)
#pragma unroll
        for (int r = 0; r < 8; ++r) {
            int row = qBase + wave * 16 + rbase + r;
            ctx[(size_t)(b * S_ + row) * D_ + h * DH_ + nt * 16 + n] = (bf16)cacc[nt][r];
        }
}

// ---------------------------------------------------------------------------
extern "C" void kernel_launch(void* const* d_in, const int* in_sizes, int n_in,
                              void* d_out, int out_size, void* d_ws, size_t ws_size,
                              hipStream_t stream) {
    const float* qs = (const float*)d_in[0];
    const float* ks = (const float*)d_in[1];
    const float* vs = (const float*)d_in[2];
    const int*  mask = (const int*)d_in[3];
    const float* Wq = (const float*)d_in[4];
    const float* bq = (const float*)d_in[5];
    const float* Wk = (const float*)d_in[6];
    const float* bk = (const float*)d_in[7];
    const float* Wv = (const float*)d_in[8];
    const float* bv = (const float*)d_in[9];
    const float* Wo = (const float*)d_in[10];
    const float* bo = (const float*)d_in[11];

    float* out  = (float*)d_out;
    float* attn = out + (size_t)B_ * S_ * D_;

    const size_t elems = (size_t)B_ * S_ * D_;   // 4,194,304
    bf16* qb  = (bf16*)d_ws;                     // 8 MB each, 32 MB total
    bf16* kb  = qb + elems;
    bf16* vb  = kb + elems;
    bf16* ctx = vb + elems;

    dim3 blk(128);
    dim3 gGemm(D_ / 64, (B_ * S_) / 64);         // (16, 64)
    ProjGemmF32ToBf16<<<gGemm, blk, 0, stream>>>(qs, Wq, bq, qb);
    ProjGemmF32ToBf16<<<gGemm, blk, 0, stream>>>(ks, Wk, bk, kb);
    ProjGemmF32ToBf16<<<gGemm, blk, 0, stream>>>(vs, Wv, bv, vb);

    dim3 gAttn(S_ / 64, H_, B_);
    AttnKernel<<<gAttn, blk, 0, stream>>>(qb, kb, vb, mask, attn, ctx);

    OutProjGemm<<<gGemm, blk, 0, stream>>>(ctx, Wo, bo, out);
}
